// FoL_20272245637651
// MI455X (gfx1250) — matrix operand order repaired and checked by reading, not compiled, not run on hardware
//
#include <hip/hip_runtime.h>

#define HWp 16384
#define Wd 128
#define Hd 128
#define Bn 16
#define Cn 256
#define C2 128
#define WCH 288
#define WCg 32
#define EPSV 1e-5f

typedef unsigned int uint32;

typedef __attribute__((ext_vector_type(16))) __bf16 v16bf;
typedef __attribute__((ext_vector_type(8)))  float  v8f;

union ABits { uint32 u[8]; v16bf v; };

__device__ __forceinline__ unsigned short f2bf(float f) {
  unsigned int u = __float_as_uint(f);
  u += 0x7FFFu + ((u >> 16) & 1u);          // round-to-nearest-even
  return (unsigned short)(u >> 16);
}
__device__ __forceinline__ float bf2f(unsigned short h) {
  return __uint_as_float(((unsigned int)h) << 16);
}

// ---------------------------------------------------------------------------
// Pre-swizzle a row-major f32 weight matrix [M][K] into bf16, laid out exactly
// as the CDNA5 16-bit A-matrix (16x32 tile, wave32):
//   lanes 0-15:  M=lane,     VGPR j: K = {0..7}+{16..23} chunk
//   lanes 16-31: M=lane-16,  VGPR j: K = {8..15}+{24..31} chunk
// Element (mt, kk, lane, j) stored contiguously -> one 32B/lane load per tile.
// ---------------------------------------------------------------------------
__global__ __launch_bounds__(256) void pack_w_kernel(const float* __restrict__ w,
                                                     uint32* __restrict__ dst,
                                                     int MT, int KC) {
  int i = blockIdx.x * 256 + threadIdx.x;
  int total = MT * KC * 256;
  if (i >= total) return;
  int j    = i & 7;
  int lane = (i >> 3) & 31;
  int kk   = (i >> 8) % KC;
  int mt   = (i >> 8) / KC;
  int m    = mt * 16 + (lane & 15);
  int g = j >> 2, jj = j & 3;
  int kb = ((lane >> 4) ? 8 : 0) + g * 16 + jj * 2;
  int k  = kk * 32 + kb;
  int K  = KC * 32;
  unsigned int lo = f2bf(w[(size_t)m * K + k]);
  unsigned int hi = f2bf(w[(size_t)m * K + k + 1]);
  dst[i] = lo | (hi << 16);
}

// ---------------------------------------------------------------------------
// 1x1 conv as GEMM via v_wmma_f32_16x16x32_bf16.
// Each wave: NB x 16 pixels (N tiles), full M (MT accumulator tiles).
// - B(kk+1) is prefetched before the WMMA chain of kk (double buffer).
// - A tiles are preloaded in groups of G so the scheduler can overlap the
//   b128 load clauses with the XDL pipe using partial loadcnt waits.
// B operand gathered per CDNA5 16-bit B layout: lanes 0-15 hold K=0..7,16..23
// (N=lane), lanes 16-31 hold K=8..15,24..31 (N=lane-16), 2 bf16 per VGPR.
// Epilogue fuses per-channel affine (+bias) and ReLU; stores bf16.
// ---------------------------------------------------------------------------
template<int M, int K, int NB, bool SRC_BF16, bool HAS_SCALE, bool RELU>
__global__ __launch_bounds__(256) void gemm1x1_kernel(
    const void* __restrict__ src, const uint32* __restrict__ apk,
    const float* __restrict__ scale, const float* __restrict__ bias,
    unsigned short* __restrict__ dst) {
  constexpr int MT = M / 16;
  constexpr int KC = K / 32;
  constexpr int G  = (MT % 4 == 0) ? 4 : 3;     // A-tile preload group
  const int wave  = threadIdx.x >> 5;
  const int lane  = threadIdx.x & 31;
  const int tile0 = (blockIdx.x * 8 + wave) * NB;
  const int p16   = tile0 << 4;
  const int b     = p16 >> 14;                  // / HWp
  const int pix   = p16 & (HWp - 1);
  const int n     = lane & 15;
  const int half  = lane >> 4;

  v8f acc[MT * NB];
  #pragma unroll
  for (int t = 0; t < MT * NB; ++t) {
    #pragma unroll
    for (int r = 0; r < 8; ++r) acc[t][r] = 0.0f;
  }

  const size_t srcBase = (size_t)b * K * HWp + (size_t)pix + n;

  auto loadB = [&](int kk, ABits* bb) {
    const int kbase = kk * 32 + half * 8;
    #pragma unroll
    for (int nb = 0; nb < NB; ++nb) {
      #pragma unroll
      for (int j = 0; j < 8; ++j) {
        unsigned int lo, hi;
        if constexpr (SRC_BF16) {
          const unsigned short* s = (const unsigned short*)src;
          lo = s[srcBase + nb * 16 + (size_t)(kbase + (j >> 2) * 16 + (j & 3) * 2) * HWp];
          hi = s[srcBase + nb * 16 + (size_t)(kbase + (j >> 2) * 16 + (j & 3) * 2 + 1) * HWp];
        } else {
          const float* s = (const float*)src;
          lo = f2bf(s[srcBase + nb * 16 + (size_t)(kbase + (j >> 2) * 16 + (j & 3) * 2) * HWp]);
          hi = f2bf(s[srcBase + nb * 16 + (size_t)(kbase + (j >> 2) * 16 + (j & 3) * 2 + 1) * HWp]);
        }
        bb[nb].u[j] = lo | (hi << 16);
      }
    }
  };

  ABits bcur[NB], bnxt[NB];
  loadB(0, bcur);
  for (int kk = 0; kk < KC; ++kk) {
    #pragma unroll
    for (int nb = 0; nb < NB; ++nb) bnxt[nb] = bcur[nb];
    if (kk + 1 < KC) loadB(kk + 1, bnxt);       // double-buffered B prefetch
    #pragma unroll
    for (int mg = 0; mg < MT; mg += G) {
      ABits aa[G];
      #pragma unroll
      for (int t = 0; t < G; ++t) {
        const uint32* ap = apk + (((size_t)((mg + t) * KC + kk) * 32 + lane) << 3);
        #pragma unroll
        for (int j = 0; j < 8; ++j) aa[t].u[j] = ap[j];
      }
      #pragma unroll
      for (int t = 0; t < G; ++t) {
        #pragma unroll
        for (int nb = 0; nb < NB; ++nb) {
          acc[(mg + t) * NB + nb] = __builtin_amdgcn_wmma_f32_16x16x32_bf16(
              false, aa[t].v, false, bcur[nb].v, (short)0,
              acc[(mg + t) * NB + nb], false, false);
        }
      }
    }
    #pragma unroll
    for (int nb = 0; nb < NB; ++nb) bcur[nb] = bnxt[nb];
  }

  // epilogue: C/D layout -> element (VGPR r, lane): m = mt*16 + half*8 + r
  #pragma unroll
  for (int mt = 0; mt < MT; ++mt) {
    #pragma unroll
    for (int r = 0; r < 8; ++r) {
      const int m = mt * 16 + half * 8 + r;
      float scl = 1.0f;
      if constexpr (HAS_SCALE) scl = scale[m];
      const float bi = bias[m];
      #pragma unroll
      for (int nb = 0; nb < NB; ++nb) {
        float v = acc[mt * NB + nb][r];
        if constexpr (HAS_SCALE) v = v * scl + bi;
        else                     v = v + bi;
        if constexpr (RELU)      v = v > 0.0f ? v : 0.0f;
        dst[(size_t)(b * M + m) * HWp + pix + nb * 16 + n] = f2bf(v);
      }
    }
  }
}

// ---------------------------------------------------------------------------
// 7x7 depthwise conv, bf16 in/out, f32 math, LDS halo tile, +affine+ReLU.
// One block = one (b, c, 16x16 spatial tile).
// ---------------------------------------------------------------------------
__global__ __launch_bounds__(256) void dw7x7_kernel(
    const unsigned short* __restrict__ src, const float* __restrict__ w2,
    const float* __restrict__ s2, const float* __restrict__ b2,
    unsigned short* __restrict__ dst) {
  __shared__ unsigned short tile[22 * 22];
  __shared__ float wsh[49];
  const int bx = blockIdx.x;
  const int tw = bx & 7, th = (bx >> 3) & 7, c = (bx >> 6) & 127, b = bx >> 13;
  const int tid = threadIdx.x;
  const size_t chBase = (size_t)(b * C2 + c) * HWp;
  for (int i = tid; i < 484; i += 256) {
    int ly = i / 22, lx = i % 22;
    int gy = th * 16 + ly - 3, gx = tw * 16 + lx - 3;
    unsigned short v = 0;
    if (gy >= 0 && gy < Hd && gx >= 0 && gx < Wd) v = src[chBase + gy * Wd + gx];
    tile[i] = v;
  }
  if (tid < 49) wsh[tid] = w2[c * 49 + tid];
  __syncthreads();
  const int py = tid >> 4, px = tid & 15;
  float a = 0.0f;
  #pragma unroll
  for (int dy = 0; dy < 7; ++dy) {
    #pragma unroll
    for (int dx = 0; dx < 7; ++dx)
      a += bf2f(tile[(py + dy) * 22 + px + dx]) * wsh[dy * 7 + dx];
  }
  float v = a * s2[c] + b2[c];
  v = v > 0.0f ? v : 0.0f;
  dst[chBase + (th * 16 + py) * Wd + tw * 16 + px] = f2bf(v);
}

// ---------------------------------------------------------------------------
// GroupNorm stats: one block per (b, wc) group; channels wc*9..wc*9+8 are
// contiguous, so the group is one contiguous run of 9*HW bf16 values.
// ---------------------------------------------------------------------------
__global__ __launch_bounds__(256) void gn_stats_kernel(
    const unsigned short* __restrict__ wkb, float2* __restrict__ stats) {
  __shared__ float ssum[256], ssq[256];
  const int grp = blockIdx.x;                    // b*32 + wc
  const size_t base = (size_t)grp * 9 * HWp;
  float s = 0.0f, q = 0.0f;
  for (int i = threadIdx.x; i < 9 * HWp; i += 256) {
    float v = bf2f(wkb[base + i]);
    s += v; q += v * v;
  }
  ssum[threadIdx.x] = s; ssq[threadIdx.x] = q;
  __syncthreads();
  for (int o = 128; o > 0; o >>= 1) {
    if (threadIdx.x < o) {
      ssum[threadIdx.x] += ssum[threadIdx.x + o];
      ssq[threadIdx.x]  += ssq[threadIdx.x + o];
    }
    __syncthreads();
  }
  if (threadIdx.x == 0) {
    float inv_n = 1.0f / (9.0f * HWp);
    float mu  = ssum[0] * inv_n;
    float var = ssq[0] * inv_n - mu * mu;
    stats[grp] = make_float2(mu, rsqrtf(var + EPSV));
  }
}

// ---------------------------------------------------------------------------
// SKA fused with GN-normalize, eval BN and residual.
// Block = (b, 16x16 spatial tile); per weight-channel wc: normalize its 9
// kernel taps once, reuse for the 8 channels with c % 32 == wc.
// x re-reads (9x) hit L2 (per-image x = 16.8 MB << 192 MB L2);
// global_prefetch_b8 hints pull the next streams while the MACs run.
// ---------------------------------------------------------------------------
__global__ __launch_bounds__(256) void ska_kernel(
    const float* __restrict__ x, const unsigned short* __restrict__ wkb,
    const float2* __restrict__ stats,
    const float* __restrict__ gng, const float* __restrict__ gnb,
    const float* __restrict__ bng, const float* __restrict__ bnb,
    const float* __restrict__ bnm, const float* __restrict__ bnv,
    float* __restrict__ out) {
  const int bx = blockIdx.x;
  const int tw = bx & 7, th = (bx >> 3) & 7, b = bx >> 6;
  const int px = threadIdx.x & 15, py = threadIdx.x >> 4;
  const int h = th * 16 + py, w = tw * 16 + px;
  const int pos = h * Wd + w;
  for (int wc = 0; wc < WCg; ++wc) {
    if (wc + 1 < WCg) {   // prefetch next weight-channel's kernel taps
      __builtin_prefetch(
          (const void*)(wkb + (size_t)(b * WCH + (wc + 1) * 9) * HWp + pos), 0, 0);
    }
    float2 st = stats[b * WCg + wc];
    float wn[9];
    #pragma unroll
    for (int k = 0; k < 9; ++k) {
      int ch = wc * 9 + k;
      float v = bf2f(wkb[(size_t)(b * WCH + ch) * HWp + pos]);
      wn[k] = (v - st.x) * st.y * gng[ch] + gnb[ch];
    }
    for (int r = 0; r < 8; ++r) {
      int c = wc + 32 * r;                     // channel c uses weight ch c%32
      const float* xc = x + (size_t)(b * Cn + c) * HWp;
      if (r + 1 < 8) {   // prefetch next channel's x line
        __builtin_prefetch((const void*)(xc + 32 * HWp + pos), 0, 0);
      }
      float a = 0.0f, xcen = 0.0f;
      #pragma unroll
      for (int dy = 0; dy < 3; ++dy) {
        int yy = h + dy - 1;
        #pragma unroll
        for (int dx = 0; dx < 3; ++dx) {
          int xx = w + dx - 1;
          float xv = (yy >= 0 && yy < Hd && xx >= 0 && xx < Wd)
                         ? xc[yy * Wd + xx] : 0.0f;
          if (dy == 1 && dx == 1) xcen = xv;
          a += xv * wn[dy * 3 + dx];
        }
      }
      float sc = bng[c] * rsqrtf(bnv[c] + EPSV);
      float v = (a - bnm[c]) * sc + bnb[c] + xcen;
      out[(size_t)(b * Cn + c) * HWp + pos] = v;
    }
  }
}

// ---------------------------------------------------------------------------
extern "C" void kernel_launch(void* const* d_in, const int* in_sizes, int n_in,
                              void* d_out, int out_size, void* d_ws, size_t ws_size,
                              hipStream_t stream) {
  (void)in_sizes; (void)n_in; (void)out_size; (void)ws_size;
  const float* x     = (const float*)d_in[0];
  const float* w1    = (const float*)d_in[1];
  const float* s1    = (const float*)d_in[2];
  const float* b1    = (const float*)d_in[3];
  const float* w2    = (const float*)d_in[4];
  const float* s2    = (const float*)d_in[5];
  const float* b2    = (const float*)d_in[6];
  const float* w3    = (const float*)d_in[7];
  const float* s3    = (const float*)d_in[8];
  const float* b3    = (const float*)d_in[9];
  const float* w4    = (const float*)d_in[10];
  const float* bias4 = (const float*)d_in[11];
  const float* gng   = (const float*)d_in[12];
  const float* gnb   = (const float*)d_in[13];
  const float* bng   = (const float*)d_in[14];
  const float* bnb   = (const float*)d_in[15];
  const float* bnm   = (const float*)d_in[16];
  const float* bnv   = (const float*)d_in[17];
  float* out = (float*)d_out;
  char* ws = (char*)d_ws;

  // workspace layout (bytes)
  uint32* w1p = (uint32*)(ws + 0);                       //  64 KB
  uint32* w3p = (uint32*)(ws + 65536);                   //  32 KB
  uint32* w4p = (uint32*)(ws + 98304);                   //  72 KB
  float2* stats = (float2*)(ws + 172032);                //   4 KB
  unsigned short* hA  = (unsigned short*)(ws + 262144);                      // 64 MB
  unsigned short* hB  = (unsigned short*)(ws + 262144 + 67108864ull);        // 64 MB
  unsigned short* wkb = (unsigned short*)(ws + 262144 + 2ull * 67108864ull); // 144 MB

  // weight pre-swizzle into WMMA A layout (bf16)
  pack_w_kernel<<<64, 256, 0, stream>>>(w1, w1p, 8, 8);    // 128x256
  pack_w_kernel<<<32, 256, 0, stream>>>(w3, w3p, 8, 4);    // 128x128
  pack_w_kernel<<<72, 256, 0, stream>>>(w4, w4p, 18, 4);   // 288x128

  // conv1 1x1 (256->128) + affine + ReLU   (NB=2: 32 pixels/wave)
  gemm1x1_kernel<128, 256, 2, false, true, true><<<1024, 256, 0, stream>>>(
      x, w1p, s1, b1, hA);
  // 7x7 depthwise + affine + ReLU
  dw7x7_kernel<<<131072, 256, 0, stream>>>(hA, w2, s2, b2, hB);
  // conv3 1x1 (128->128) + affine + ReLU (reuse hA for h3)
  gemm1x1_kernel<128, 128, 2, true, true, true><<<1024, 256, 0, stream>>>(
      hB, w3p, s3, b3, hA);
  // conv4 1x1 (128->288) + bias -> wk (bf16)   (NB=1: 288 acc VGPRs otherwise)
  gemm1x1_kernel<288, 128, 1, true, false, false><<<2048, 256, 0, stream>>>(
      hA, w4p, nullptr, bias4, wkb);
  // GroupNorm stats per (b, wc)
  gn_stats_kernel<<<512, 256, 0, stream>>>(wkb, stats);
  // SKA + GN-apply + BN + residual
  ska_kernel<<<1024, 256, 0, stream>>>(x, wkb, stats, gng, gnb,
                                       bng, bnb, bnm, bnv, out);
}
